// Attention_45655502357001
// MI455X (gfx1250) — compile-verified, matrix operand order checked by reading, not run on hardware
//
#include <hip/hip_runtime.h>

typedef __bf16 bf16;
typedef __attribute__((ext_vector_type(4)))  __bf16 v4bf;
typedef __attribute__((ext_vector_type(8)))  __bf16 v8bf;
typedef __attribute__((ext_vector_type(16))) __bf16 v16bf;
typedef __attribute__((ext_vector_type(4)))  float  v4f;
typedef __attribute__((ext_vector_type(8)))  float  v8f;

#define B_   2
#define L_   2048
#define HID_ 2048
#define H_   16
#define KV_  4
#define D_   128
#define WIN_ 512

union Frag16 { v16bf v; v8bf h[2]; };

// Load a 16x32 bf16 WMMA A/B fragment. `base` is a row-major matrix with
// leading dim `ld` (elements). Per ISA 7.12.2: lanes 0-15 hold K=k0+0..7
// (elems 0-7) and K=k0+16..23 (elems 8-15) of row row0+lane; lanes 16-31
// hold K=k0+8..15 and K=k0+24..31.
__device__ __forceinline__ v16bf load_frag(const bf16* base, int ld, int row0,
                                           int k0, int lane) {
  const int half = lane >> 4;
  const int r    = lane & 15;
  const bf16* rp = base + (size_t)(row0 + r) * ld + (k0 + half * 8);
  Frag16 f;
  f.h[0] = *(const v8bf*)rp;
  f.h[1] = *(const v8bf*)(rp + 16);
  return f.v;
}

__device__ __forceinline__ v8f wmma_bf16(v16bf a, v16bf b, v8f c) {
  return __builtin_amdgcn_wmma_f32_16x16x32_bf16(
      /*neg_a=*/false, a, /*neg_b=*/false, b,
      /*c_mod=*/(short)0, c, /*reuse_a=*/false, /*reuse_b=*/false);
}

// Async 16-byte copy global -> LDS (per lane), tracked by ASYNCcnt.
// LDS flat addresses carry the LDS byte offset in the low 32 bits.
__device__ __forceinline__ void async_copy_b128(const bf16* gsrc,
                                                const bf16* ldst) {
  unsigned lofs = (unsigned)(size_t)ldst;
  asm volatile("global_load_async_to_lds_b128 %0, %1, off"
               :: "v"(lofs), "v"(gsrc) : "memory");
}

// ---------------------------------------------------------------- cast f32->bf16
__global__ void cast_bf16_kernel(const float* __restrict__ in,
                                 bf16* __restrict__ out, int n4) {
  int i = blockIdx.x * blockDim.x + threadIdx.x;
  if (i >= n4) return;
  v4f x = ((const v4f*)in)[i];
  v4bf y;
  y[0] = (bf16)x[0]; y[1] = (bf16)x[1]; y[2] = (bf16)x[2]; y[3] = (bf16)x[3];
  ((v4bf*)out)[i] = y;
}

// --------------------------------------------- C[M,N](f32) = A[M,K] * B[N,K]^T
// One block = 128x64 tile: 8 waves x 16 rows, all sharing one 64-column B
// tile. The 64x32 B k-slice (4KB) is double-buffered in LDS via
// global_load_async_to_lds_b128 (1 async b128 per wave per k-step), and read
// back as WMMA fragments with ds_load_b128. A fragments come straight from
// global (each wave owns distinct rows).
__global__ void __launch_bounds__(256)
gemm_nt_bf16(const bf16* __restrict__ A, const bf16* __restrict__ Bm,
             float* __restrict__ C, int M, int N, int K) {
  const int lane = threadIdx.x & 31;
  const int wid  = threadIdx.x >> 5;                 // 0..7
  const int ntiles = N >> 6;                         // N / 64
  const int n0 = (blockIdx.x % ntiles) << 6;
  const int m0 = (blockIdx.x / ntiles) * 128 + wid * 16;

  __shared__ bf16 sB[2][64 * 32];                    // 2 x 4KB

  // This thread's async-copy unit: one b128 (8 bf16) of the 64x32 B slice.
  const int unit = wid * 32 + lane;                  // 0..255
  const int brow = unit >> 2;                        // 0..63
  const int bseg = (unit & 3) * 8;                   // 0,8,16,24
  const bf16* gB  = Bm + (size_t)(n0 + brow) * K + bseg;
  bf16*       lB0 = &sB[0][brow * 32 + bseg];
  bf16*       lB1 = &sB[1][brow * 32 + bseg];

  async_copy_b128(gB, lB0);                          // prime buffer 0

  v8f acc0 = {}, acc1 = {}, acc2 = {}, acc3 = {};
  for (int k0 = 0; k0 < K; k0 += 32) {
    const int  cur  = (k0 >> 5) & 1;
    const bool more = (k0 + 32) < K;
    if (more) {
      async_copy_b128(gB + k0 + 32, cur ? lB0 : lB1);
      asm volatile("s_wait_asynccnt 0x1" ::: "memory");  // this tile done
    } else {
      asm volatile("s_wait_asynccnt 0x0" ::: "memory");
    }
    __syncthreads();                                 // tile visible to block

    v16bf a  = load_frag(A, K, m0, k0, lane);
    const bf16* bt = sB[cur];
    v16bf b0 = load_frag(bt, 32,  0, 0, lane);
    v16bf b1 = load_frag(bt, 32, 16, 0, lane);
    v16bf b2 = load_frag(bt, 32, 32, 0, lane);
    v16bf b3 = load_frag(bt, 32, 48, 0, lane);
    acc0 = wmma_bf16(a, b0, acc0);
    acc1 = wmma_bf16(a, b1, acc1);
    acc2 = wmma_bf16(a, b2, acc2);
    acc3 = wmma_bf16(a, b3, acc3);

    __syncthreads();                                 // done reading cur buffer
  }

  const int half = lane >> 4, c = lane & 15;
  float* Cr = C + (size_t)(m0 + half * 8) * N + n0 + c;
  for (int r = 0; r < 8; ++r) {
    Cr[0] = acc0[r]; Cr[16] = acc1[r]; Cr[32] = acc2[r]; Cr[48] = acc3[r];
    Cr += N;
  }
}

// ---------------------------------- fused RMSNorm + RoPE, write [B,NH,L,D] bf16
// One wave per (token, head). Lane owns 4 consecutive d's.
__global__ void norm_rope_kernel(const float* __restrict__ inp,   // [B*L, NH*D]
                                 const float* __restrict__ w,     // [D]
                                 const float* __restrict__ cosT,  // [L, D]
                                 const float* __restrict__ sinT,  // [L, D]
                                 bf16* __restrict__ outp,         // [B, NH, L, D]
                                 int NH) {
  const int lane = threadIdx.x & 31;
  const int wave = (int)((blockIdx.x * blockDim.x + threadIdx.x) >> 5);
  const int tok  = wave / NH;
  const int head = wave % NH;
  if (tok >= B_ * L_) return;
  const int l  = tok & (L_ - 1);
  const int d0 = lane * 4;

  const float* xr = inp + (size_t)tok * (NH * D_) + head * D_ + d0;
  v4f x = *(const v4f*)xr;
  float ss = x[0]*x[0] + x[1]*x[1] + x[2]*x[2] + x[3]*x[3];
  for (int o = 1; o < 32; o <<= 1) ss += __shfl_xor(ss, o, 32);
  const float rn = rsqrtf(ss * (1.0f / D_) + 1e-6f);

  v4f xw = *(const v4f*)(w + d0);
  const float* cr = cosT + (size_t)l * D_ + d0;
  const float* sr = sinT + (size_t)l * D_ + d0;
  const float sgn = (d0 < 64) ? -1.0f : 1.0f;   // rotate_half sign

  float xn[4], o4[4];
  for (int i = 0; i < 4; ++i) xn[i] = x[i] * rn * xw[i];
  for (int i = 0; i < 4; ++i) {
    float part = __shfl_xor(xn[i], 16, 32);     // value at d ^ 64
    o4[i] = xn[i] * cr[i] + sgn * part * sr[i];
  }
  bf16* orow = outp + (((size_t)(tok / L_) * NH + head) * L_ + l) * (size_t)D_ + d0;
  v4bf y; for (int i = 0; i < 4; ++i) y[i] = (bf16)o4[i];
  *(v4bf*)orow = y;
}

// ------------------------- V: [B*L, KV*D] f32 -> [B, KV, D, L] bf16 (transposed)
__global__ void vcast_transpose_kernel(const float* __restrict__ vf,
                                       bf16* __restrict__ vt) {
  int i = blockIdx.x * blockDim.x + threadIdx.x;  // over B*L*KV*D
  if (i >= B_ * L_ * KV_ * D_) return;
  int d   = i & (D_ - 1);
  int t   = i >> 7;          // / D_
  int kv  = t & (KV_ - 1);
  int tok = t >> 2;          // / KV_
  int l   = tok & (L_ - 1);
  int b   = tok >> 11;       // / L_
  vt[(((size_t)b * KV_ + kv) * D_ + d) * (size_t)L_ + l] = (bf16)vf[i];
}

// ------------------------------------------ sliding-window flash attention
// One wave per (b, h, 16-query tile). Q kept resident as 4 A-fragments;
// keys processed in 32-wide chunks restricted to the +-512 window.
__global__ void __launch_bounds__(128)
attn_kernel(const bf16* __restrict__ Qa,   // [B, H, L, D]
            const bf16* __restrict__ Ka,   // [B, KV, L, D]
            const bf16* __restrict__ Vt,   // [B, KV, D, L]
            bf16* __restrict__ Oa) {       // [B, L, H*D]
  const int lane = threadIdx.x & 31;
  const int wid  = threadIdx.x >> 5;
  const int wave = (int)((blockIdx.x * blockDim.x + threadIdx.x) >> 5);
  const int QT = L_ / 16;
  const int qt = wave % QT;
  const int h  = (wave / QT) % H_;
  const int b  = wave / (QT * H_);
  const int kv = h / (H_ / KV_);
  const int q0 = qt * 16;
  const int half = lane >> 4, c = lane & 15;

  __shared__ bf16 sP[4][16 * 32];          // per-wave P tile (C-layout -> A-layout)
  bf16* myP = sP[wid];

  const bf16* Qb = Qa + (((size_t)b * H_ + h) * L_ + q0) * (size_t)D_;
  const bf16* Kb = Ka + (((size_t)b * KV_ + kv) * L_) * (size_t)D_;
  const bf16* Vb = Vt + (((size_t)b * KV_ + kv) * D_) * (size_t)L_;

  v16bf qa0 = load_frag(Qb, D_, 0,  0, lane);
  v16bf qa1 = load_frag(Qb, D_, 0, 32, lane);
  v16bf qa2 = load_frag(Qb, D_, 0, 64, lane);
  v16bf qa3 = load_frag(Qb, D_, 0, 96, lane);

  v8f acc[8] = {};                          // 16x128 O tile
  float mx[8], sm[8];
  for (int r = 0; r < 8; ++r) { mx[r] = -1e30f; sm[r] = 0.0f; }

  int ks = q0 - WIN_; if (ks < 0) ks = 0; ks &= ~31;
  const int ke = (q0 + 16 + WIN_ < L_) ? (q0 + 16 + WIN_) : L_;
  const float scale = 0.08838834764831845f; // 1/sqrt(128)

  for (int kc = ks; kc < ke; kc += 32) {
    if (kc + 32 < ke) {                     // stream-prefetch next key chunk
      __builtin_prefetch(Kb + (size_t)(kc + 32 + lane) * D_, 0, 1);
      __builtin_prefetch(Vb + (size_t)(lane * 4) * L_ + kc + 32, 0, 1);
    }
    // S tile: 16 queries x 32 keys, contraction over D=128
    v8f s0 = {}, s1 = {};
    {
      v16bf b0, b1;
      b0 = load_frag(Kb, D_, kc,      0, lane); b1 = load_frag(Kb, D_, kc + 16,  0, lane);
      s0 = wmma_bf16(qa0, b0, s0);              s1 = wmma_bf16(qa0, b1, s1);
      b0 = load_frag(Kb, D_, kc,     32, lane); b1 = load_frag(Kb, D_, kc + 16, 32, lane);
      s0 = wmma_bf16(qa1, b0, s0);              s1 = wmma_bf16(qa1, b1, s1);
      b0 = load_frag(Kb, D_, kc,     64, lane); b1 = load_frag(Kb, D_, kc + 16, 64, lane);
      s0 = wmma_bf16(qa2, b0, s0);              s1 = wmma_bf16(qa2, b1, s1);
      b0 = load_frag(Kb, D_, kc,     96, lane); b1 = load_frag(Kb, D_, kc + 16, 96, lane);
      s0 = wmma_bf16(qa3, b0, s0);              s1 = wmma_bf16(qa3, b1, s1);
    }
    // mask + scale + online softmax (row reductions across 16-lane halves)
    const int kp0 = kc + c, kp1 = kc + 16 + c;
    for (int r = 0; r < 8; ++r) {
      const int qp = q0 + r + half * 8;
      int dd0 = qp - kp0; if (dd0 < 0) dd0 = -dd0;
      int dd1 = qp - kp1; if (dd1 < 0) dd1 = -dd1;
      float v0 = (dd0 <= WIN_) ? s0[r] * scale : -1e30f;
      float v1 = (dd1 <= WIN_) ? s1[r] * scale : -1e30f;
      float cm = fmaxf(v0, v1);
      for (int o = 1; o < 16; o <<= 1) cm = fmaxf(cm, __shfl_xor(cm, o, 32));
      const float nm   = fmaxf(mx[r], cm);
      const float corr = __expf(mx[r] - nm);
      mx[r] = nm;
      const float p0 = __expf(v0 - nm);
      const float p1 = __expf(v1 - nm);
      float ps = p0 + p1;
      for (int o = 1; o < 16; o <<= 1) ps += __shfl_xor(ps, o, 32);
      sm[r] = sm[r] * corr + ps;
      for (int jt = 0; jt < 8; ++jt) acc[jt][r] *= corr;
      const int row = r + half * 8;
      myP[row * 32 + c]      = (bf16)p0;
      myP[row * 32 + 16 + c] = (bf16)p1;
    }
    asm volatile("s_wait_dscnt 0" ::: "memory");
    // reload P in A-operand layout
    Frag16 pf;
    pf.h[0] = *(const v8bf*)(myP + c * 32 + half * 8);
    pf.h[1] = *(const v8bf*)(myP + c * 32 + 16 + half * 8);
    // O += P * V  (V^T rows are contiguous along keys)
    for (int jt = 0; jt < 8; ++jt) {
      v16bf vfrag = load_frag(Vb, L_, jt * 16, kc, lane);
      acc[jt] = wmma_bf16(pf.v, vfrag, acc[jt]);
    }
  }
  // epilogue: normalize and store bf16 [B, L, H*D]
  bf16* Ob = Oa + ((size_t)b * L_ + q0 + half * 8) * (size_t)(H_ * D_) + h * D_ + c;
  for (int r = 0; r < 8; ++r) {
    const float inv = 1.0f / sm[r];
    for (int jt = 0; jt < 8; ++jt)
      Ob[(size_t)r * (H_ * D_) + jt * 16] = (bf16)(acc[jt][r] * inv);
  }
}

// ---------------------------------------------------------------------- host
extern "C" void kernel_launch(void* const* d_in, const int* in_sizes, int n_in,
                              void* d_out, int out_size, void* d_ws, size_t ws_size,
                              hipStream_t stream) {
  (void)in_sizes; (void)n_in; (void)out_size; (void)ws_size;
  const float* x    = (const float*)d_in[0];
  const float* cosT = (const float*)d_in[1];
  const float* sinT = (const float*)d_in[2];
  const float* Wq   = (const float*)d_in[3];
  const float* Wk   = (const float*)d_in[4];
  const float* Wv   = (const float*)d_in[5];
  const float* Wo   = (const float*)d_in[6];
  const float* qw   = (const float*)d_in[7];
  const float* kw   = (const float*)d_in[8];
  float* out = (float*)d_out;

  char* ws = (char*)d_ws;
  size_t off = 0;
  auto alloc = [&](size_t bytes) -> void* {
    void* p = ws + off;
    off += (bytes + 255) & ~(size_t)255;
    return p;
  };
  const size_t TOK = (size_t)B_ * L_;                 // 4096 tokens
  bf16*  xb  = (bf16*) alloc(TOK * HID_ * 2);         // x bf16
  bf16*  wqb = (bf16*) alloc((size_t)HID_ * HID_ * 2);
  bf16*  wkb = (bf16*) alloc((size_t)KV_ * D_ * HID_ * 2);
  bf16*  wvb = (bf16*) alloc((size_t)KV_ * D_ * HID_ * 2);
  bf16*  wob = (bf16*) alloc((size_t)HID_ * HID_ * 2);
  float* qf  = (float*)alloc(TOK * HID_ * 4);         // Q proj f32
  float* kf  = (float*)alloc(TOK * KV_ * D_ * 4);
  float* vf  = (float*)alloc(TOK * KV_ * D_ * 4);
  bf16*  qa  = (bf16*) alloc(TOK * HID_ * 2);         // [B,H,L,D]
  bf16*  ka  = (bf16*) alloc(TOK * KV_ * D_ * 2);     // [B,KV,L,D]
  bf16*  vt  = (bf16*) alloc(TOK * KV_ * D_ * 2);     // [B,KV,D,L]
  bf16*  ao  = (bf16*) alloc(TOK * HID_ * 2);         // attention out [B,L,H*D]

  auto cast = [&](const float* src, bf16* dst, size_t n) {
    int n4 = (int)(n / 4);
    cast_bf16_kernel<<<(n4 + 255) / 256, 256, 0, stream>>>(src, dst, n4);
  };
  cast(x,  xb,  TOK * HID_);
  cast(Wq, wqb, (size_t)HID_ * HID_);
  cast(Wk, wkb, (size_t)KV_ * D_ * HID_);
  cast(Wv, wvb, (size_t)KV_ * D_ * HID_);
  cast(Wo, wob, (size_t)HID_ * HID_);

  auto gemm = [&](const bf16* A, const bf16* Bm, float* C, int M, int N, int K) {
    int blocks = (M / 128) * (N / 64);
    gemm_nt_bf16<<<blocks, 256, 0, stream>>>(A, Bm, C, M, N, K);
  };
  gemm(xb, wqb, qf, (int)TOK, H_ * D_,  HID_);
  gemm(xb, wkb, kf, (int)TOK, KV_ * D_, HID_);
  gemm(xb, wvb, vf, (int)TOK, KV_ * D_, HID_);

  {
    int waves = (int)TOK * H_;
    norm_rope_kernel<<<(waves * 32 + 255) / 256, 256, 0, stream>>>(qf, qw, cosT, sinT, qa, H_);
  }
  {
    int waves = (int)TOK * KV_;
    norm_rope_kernel<<<(waves * 32 + 255) / 256, 256, 0, stream>>>(kf, kw, cosT, sinT, ka, KV_);
  }
  {
    int n = (int)(TOK * KV_ * D_);
    vcast_transpose_kernel<<<(n + 255) / 256, 256, 0, stream>>>(vf, vt);
  }
  {
    int waves = B_ * H_ * (L_ / 16);
    attn_kernel<<<(waves * 32 + 127) / 128, 128, 0, stream>>>(qa, ka, vt, ao);
  }
  gemm(ao, wob, out, (int)TOK, HID_, H_ * D_);
}
